// TargetAssigner_5377299054974
// MI455X (gfx1250) — 8-bit matrix operand order repaired and checked by reading, compiled, not run on hardware
//
#include <hip/hip_runtime.h>
#include <hip/hip_bf16.h>

#define BB 4
#define NN 32768
#define BOX_COUNT 40
#define NUM_ANCHORS 3
#define M_NEG 512

typedef __attribute__((ext_vector_type(8))) int   v8i;
typedef __attribute__((ext_vector_type(8))) float v8f;

__global__ void ta_zero_mask(unsigned char* __restrict__ mask) {
    mask[blockIdx.x * 256 + threadIdx.x] = 0;
}

__global__ void ta_scatter_neg(const int* __restrict__ neg_inds,
                               unsigned char* __restrict__ mask) {
    int t = blockIdx.x * 256 + threadIdx.x;
    if (t < BB * M_NEG) mask[neg_inds[t]] = 1;
}

__global__ __launch_bounds__(256) void ta_assign(
    const float* __restrict__ boxes,
    const float* __restrict__ keypoints,
    const int*   __restrict__ class_ids,
    const float* __restrict__ anchor_sizes,
    const float* __restrict__ anchor_radii,
    const unsigned char* __restrict__ negmask,
    float* __restrict__ out_cls,
    float* __restrict__ out_reg)
{
    __shared__ float4 s_boxc[BOX_COUNT];
    __shared__ int    s_bcls[BOX_COUNT];
    __shared__ float  s_reg[NUM_ANCHORS][8];
    __shared__ float  s_m[8][32][4];

    const int tid  = threadIdx.x;
    const int lane = tid & 31;
    const int w    = tid >> 5;
    const int lhi  = lane >> 4;
    const int ncol = lane & 15;

    const int blocksPerBatch = NN / 256;
    const int b = blockIdx.x / blocksPerBatch;
    const int n = (blockIdx.x % blocksPerBatch) * 256 + tid;

    if (tid < BOX_COUNT) {
        const int g = b * BOX_COUNT + tid;
        const float* bp = boxes + (size_t)g * 7;
        const int cid = class_ids[g];
        const float r = anchor_radii[cid];
        s_boxc[tid] = make_float4(bp[0], bp[1], bp[2], r * r);
        s_bcls[tid] = cid;
    }
    if (tid < NUM_ANCHORS) {
        const int c = tid;
        const float* bp = boxes + (size_t)c * 7;
        s_reg[c][0] = bp[0];
        s_reg[c][1] = bp[1];
        s_reg[c][2] = bp[2];
        #pragma unroll
        for (int j = 0; j < 3; ++j) {
            const float a = anchor_sizes[c * 3 + j];
            s_reg[c][3 + j] = (bp[3 + j] - a) / a;
        }
        s_reg[c][6] = bp[6];
    }
    __syncthreads();

    const float* kp = keypoints + ((size_t)b * NN + n) * 3;
    const float kx = kp[0], ky = kp[1], kz = kp[2];

    v8i Bm;
    #pragma unroll
    for (int v = 0; v < 8; ++v) {
        unsigned word = 0u;
        #pragma unroll
        for (int bb = 0; bb < 4; ++bb) {
            const int e = v * 4 + bb;
            const int k = (e & 15) + (lhi << 4) + ((e >> 4) << 5);
            if (k < BOX_COUNT && s_bcls[k] == ncol) word |= (0x38u << (bb * 8));
        }
        Bm[v] = (int)word;
    }

    #pragma unroll
    for (int t = 0; t < 2; ++t) {
        const int src = ncol + 16 * t;
        const float rx = __shfl(kx, src, 32);
        const float ry = __shfl(ky, src, 32);
        const float rz = __shfl(kz, src, 32);

        v8i Am;
        #pragma unroll
        for (int v = 0; v < 8; ++v) {
            unsigned word = 0u;
            #pragma unroll
            for (int bb = 0; bb < 4; ++bb) {
                const int k = bb + ((v & 3) << 2) + (lhi << 4)
                            + ((v >> 2) << 5);
                if (k < BOX_COUNT) {
                    const float4 bc = s_boxc[k];
                    const float dx = bc.x - rx;
                    const float dy = bc.y - ry;
                    const float dz = bc.z - rz;
                    const float d2 = dx * dx + dy * dy + dz * dz;
                    if (d2 < bc.w) word |= (0x38u << (bb * 8));
                }
            }
            Am[v] = (int)word;
        }

        v8f cacc = {};
        cacc = __builtin_amdgcn_wmma_f32_16x16x64_fp8_fp8(
            Am, Bm, (short)0, cacc, false, false);

        if (ncol < NUM_ANCHORS) {
            #pragma unroll
            for (int r = 0; r < 8; ++r)
                s_m[w][t * 16 + r + lhi * 8][ncol] = cacc[r];
        }
    }
    __syncthreads();

    const float m0 = s_m[w][lane][0];
    const float m1 = s_m[w][lane][1];
    const float m2 = s_m[w][lane][2];
    const bool p0 = m0 > 0.5f;
    const bool p1 = m1 > 0.5f;
    const bool p2 = m2 > 0.5f;
    const bool any_pos = p0 | p1 | p2;
    const bool neg = negmask[n] != 0;

    {
        const size_t base = ((size_t)b * NN + n) * 5;
        out_cls[base + 0] = p0 ? 1.0f : 0.0f;
        out_cls[base + 1] = p1 ? 1.0f : 0.0f;
        out_cls[base + 2] = p2 ? 1.0f : 0.0f;
        out_cls[base + 3] = (neg && !any_pos) ? 1.0f : 0.0f;
        out_cls[base + 4] = (!neg && !any_pos) ? 1.0f : 0.0f;
    }

    {
        const size_t base = ((size_t)b * NN + n) * 28;
        const bool pc[NUM_ANCHORS] = {p0, p1, p2};
        #pragma unroll
        for (int c = 0; c < NUM_ANCHORS; ++c) {
            const bool m = pc[c];
            out_reg[base + c * 7 + 0] = m ? (s_reg[c][0] - kx) : 0.0f;
            out_reg[base + c * 7 + 1] = m ? (s_reg[c][1] - ky) : 0.0f;
            out_reg[base + c * 7 + 2] = m ? (s_reg[c][2] - kz) : 0.0f;
            out_reg[base + c * 7 + 3] = m ? s_reg[c][3] : 0.0f;
            out_reg[base + c * 7 + 4] = m ? s_reg[c][4] : 0.0f;
            out_reg[base + c * 7 + 5] = m ? s_reg[c][5] : 0.0f;
            out_reg[base + c * 7 + 6] = m ? s_reg[c][6] : 0.0f;
        }
        #pragma unroll
        for (int j = 0; j < 7; ++j)
            out_reg[base + 21 + j] = 0.0f;
    }
}

extern "C" void kernel_launch(void* const* d_in, const int* in_sizes, int n_in,
                              void* d_out, int out_size, void* d_ws, size_t ws_size,
                              hipStream_t stream) {
    (void)in_sizes; (void)n_in; (void)out_size; (void)ws_size;
    const float* boxes        = (const float*)d_in[0];
    const float* keypoints    = (const float*)d_in[1];
    const int*   class_ids    = (const int*)d_in[2];
    const int*   neg_inds     = (const int*)d_in[3];
    const float* anchor_sizes = (const float*)d_in[4];
    const float* anchor_radii = (const float*)d_in[5];

    unsigned char* mask = (unsigned char*)d_ws;
    float* out_cls = (float*)d_out;
    float* out_reg = (float*)d_out + (size_t)BB * NN * 5;

    ta_zero_mask<<<NN / 256, 256, 0, stream>>>(mask);
    ta_scatter_neg<<<(BB * M_NEG + 255) / 256, 256, 0, stream>>>(neg_inds, mask);
    ta_assign<<<(BB * NN) / 256, 256, 0, stream>>>(
        boxes, keypoints, class_ids, anchor_sizes, anchor_radii, mask,
        out_cls, out_reg);
}
